// QuantizedLinear_30107720744999
// MI455X (gfx1250) — 8-bit matrix operand order repaired and checked by reading, compiled, not run on hardware
//
#include <hip/hip_runtime.h>
#include <stdint.h>

typedef __attribute__((ext_vector_type(8))) int v8i;

#define M_DIM 8192
#define K_DIM 4096
#define N_DIM 4096

#define XQ_BYTES ((size_t)M_DIM * (size_t)K_DIM)
#define WQ_BYTES ((size_t)N_DIM * (size_t)K_DIM)

#if __has_builtin(__builtin_amdgcn_global_load_async_to_lds_b128) && \
    __has_builtin(__builtin_amdgcn_s_wait_asynccnt)
#define USE_ASYNC_COPY 1
#else
#define USE_ASYNC_COPY 0
#endif

typedef int v4si __attribute__((vector_size(16)));
typedef __attribute__((address_space(1))) v4si* gv4_p;
typedef __attribute__((address_space(3))) v4si* lv4_p;

__device__ __forceinline__ void copy16_g2lds(const int8_t* __restrict__ g,
                                             unsigned char* __restrict__ l) {
#if USE_ASYNC_COPY
    __builtin_amdgcn_global_load_async_to_lds_b128((gv4_p)g, (lv4_p)l, 0, 0);
#else
    *(int4*)l = *(const int4*)g;
#endif
}

__device__ __forceinline__ unsigned encf(float f) {
    unsigned u = __float_as_uint(f);
    return (u & 0x80000000u) ? ~u : (u | 0x80000000u);
}
__device__ __forceinline__ float decf(unsigned e) {
    unsigned u = (e & 0x80000000u) ? (e & 0x7fffffffu) : ~e;
    return __uint_as_float(u);
}

__global__ void init_stats(unsigned* stats) {
    stats[0] = 0xFFFFFFFFu;
    stats[1] = 0u;
    stats[2] = 0u;
    stats[3] = 0u;
}

__global__ __launch_bounds__(256) void minmax_kernel(const float4* __restrict__ x4,
                                                     int n4, unsigned* __restrict__ stats) {
    __shared__ unsigned smin[256];
    __shared__ unsigned smax[256];
    unsigned emin = 0xFFFFFFFFu, emax = 0u;
    int i = blockIdx.x * 256 + threadIdx.x;
    const int stride = gridDim.x * 256;
    for (; i < n4; i += stride) {
        float4 v = x4[i];
        unsigned e0 = encf(v.x), e1 = encf(v.y), e2 = encf(v.z), e3 = encf(v.w);
        emin = e0 < emin ? e0 : emin;  emax = e0 > emax ? e0 : emax;
        emin = e1 < emin ? e1 : emin;  emax = e1 > emax ? e1 : emax;
        emin = e2 < emin ? e2 : emin;  emax = e2 > emax ? e2 : emax;
        emin = e3 < emin ? e3 : emin;  emax = e3 > emax ? e3 : emax;
    }
    const int t = threadIdx.x;
    smin[t] = emin; smax[t] = emax;
    __syncthreads();
    for (int s = 128; s > 0; s >>= 1) {
        if (t < s) {
            smin[t] = smin[t + s] < smin[t] ? smin[t + s] : smin[t];
            smax[t] = smax[t + s] > smax[t] ? smax[t + s] : smax[t];
        }
        __syncthreads();
    }
    if (t == 0) {
        atomicMin(&stats[0], smin[0]);
        atomicMax(&stats[1], smax[0]);
    }
}

__global__ void compute_params(unsigned* stats) {
    float xmin = decf(stats[0]);
    float xmax = decf(stats[1]);
    float a_scale = (xmax - xmin) / 255.0f;
    float zp = 0.0f;
    if (a_scale != 0.0f)
        zp = fminf(127.0f, fmaxf(-128.0f, rintf(-128.0f - xmin / a_scale)));
    stats[2] = __float_as_uint(a_scale);
    stats[3] = __float_as_uint(zp);
}

__global__ __launch_bounds__(256) void quant_w_kernel(const float* __restrict__ w,
                                                      int8_t* __restrict__ wq,
                                                      float* __restrict__ wscale,
                                                      int* __restrict__ wrowsum) {
    __shared__ float redf[256];
    __shared__ int   redi[256];
    const int o = blockIdx.x;
    const int t = threadIdx.x;
    const float* row = w + (size_t)o * K_DIM;

    float amax = 0.0f;
    for (int j = t; j < K_DIM; j += 256)
        amax = fmaxf(amax, fabsf(row[j]));
    redf[t] = amax;
    __syncthreads();
    for (int s = 128; s > 0; s >>= 1) {
        if (t < s) redf[t] = fmaxf(redf[t], redf[t + s]);
        __syncthreads();
    }
    const float scale  = redf[0] / 127.0f;
    const float s_safe = (scale == 0.0f) ? 1.0f : scale;

    int acc = 0;
    int8_t* qrow = wq + (size_t)o * K_DIM;
    for (int j = t; j < K_DIM; j += 256) {
        float q = fminf(127.0f, fmaxf(-128.0f, rintf(row[j] / s_safe)));
        int qi = (int)q;
        qrow[j] = (int8_t)qi;
        acc += qi;
    }
    redi[t] = acc;
    __syncthreads();
    for (int s = 128; s > 0; s >>= 1) {
        if (t < s) redi[t] += redi[t + s];
        __syncthreads();
    }
    if (t == 0) { wscale[o] = scale; wrowsum[o] = redi[0]; }
}

__global__ __launch_bounds__(256) void quant_x_kernel(const float4* __restrict__ x4,
                                                      int* __restrict__ xq4,
                                                      const unsigned* __restrict__ stats,
                                                      int n4) {
    const float a_scale = __uint_as_float(stats[2]);
    const float zp      = __uint_as_float(stats[3]);
    int i = blockIdx.x * 256 + threadIdx.x;
    const int stride = gridDim.x * 256;
    if (a_scale != 0.0f) {
        for (; i < n4; i += stride) {
            float4 v = x4[i];
            int q0 = (int)fminf(127.0f, fmaxf(-128.0f, rintf(v.x / a_scale + zp)));
            int q1 = (int)fminf(127.0f, fmaxf(-128.0f, rintf(v.y / a_scale + zp)));
            int q2 = (int)fminf(127.0f, fmaxf(-128.0f, rintf(v.z / a_scale + zp)));
            int q3 = (int)fminf(127.0f, fmaxf(-128.0f, rintf(v.w / a_scale + zp)));
            xq4[i] = (q0 & 255) | ((q1 & 255) << 8) | ((q2 & 255) << 16) | ((q3 & 255) << 24);
        }
    } else {
        for (; i < n4; i += stride) xq4[i] = 0;
    }
}

__global__ __launch_bounds__(256) void gemm_i8_wmma(const int8_t* __restrict__ xq,
                                                    const int8_t* __restrict__ wq,
                                                    const float* __restrict__ wscale,
                                                    const int* __restrict__ wrowsum,
                                                    const float* __restrict__ bias,
                                                    const unsigned* __restrict__ stats,
                                                    float* __restrict__ out) {
    constexpr int BM = 128, BN = 256;
    constexpr int LDSS = 80;
    constexpr int ABUF = BM * LDSS;
    constexpr int BBUF = BN * LDSS;
    __shared__ alignas(16) unsigned char sA[2 * ABUF];
    __shared__ alignas(16) unsigned char sB[2 * BBUF];

    const int tid   = threadIdx.x;
    const int lane  = tid & 31;
    const int wid   = tid >> 5;
    const int waveM = wid & 1;
    const int waveN = wid >> 1;
    const int half  = lane >> 4;
    const int l16   = lane & 15;

    const int blockN0 = blockIdx.x * BN;
    const int blockM0 = blockIdx.y * BM;

    const float a_scale = __uint_as_float(stats[2]);
    const float zp      = __uint_as_float(stats[3]);

    union Acc { v8i v; int s[8]; };
    Acc acc[4][4];
    #pragma unroll
    for (int i = 0; i < 4; i++)
        #pragma unroll
        for (int j = 0; j < 4; j++)
            #pragma unroll
            for (int r = 0; r < 8; r++)
                acc[i][j].s[r] = 0;

    const int mbase = waveM * 64;
    const int nbase = waveN * 64;

    auto issue_tile = [&](int buf, int k0) {
        unsigned char* dA = &sA[buf * ABUF];
        unsigned char* dB = &sB[buf * BBUF];
        #pragma unroll
        for (int i = 0; i < 2; i++) {
            const int c   = tid + i * 256;
            const int row = c >> 2;
            const int col = (c & 3) * 16;
            copy16_g2lds(&xq[(size_t)(blockM0 + row) * K_DIM + k0 + col],
                         &dA[row * LDSS + col]);
        }
        #pragma unroll
        for (int i = 0; i < 4; i++) {
            const int c   = tid + i * 256;
            const int row = c >> 2;
            const int col = (c & 3) * 16;
            copy16_g2lds(&wq[(size_t)(blockN0 + row) * K_DIM + k0 + col],
                         &dB[row * LDSS + col]);
        }
    };

    auto compute_tile = [&](int buf) {
        const unsigned char* cA = &sA[buf * ABUF];
        const unsigned char* cB = &sB[buf * BBUF];
        union FA { v8i v; unsigned long long u[4]; };
        union FB { v8i v; int4 q[2]; };
        FA fa[4];
        FB fb[4];
        #pragma unroll
        for (int mt = 0; mt < 4; mt++) {
            const int base = (mbase + mt * 16 + l16) * LDSS + 16 * half;
            #pragma unroll
            for (int j = 0; j < 4; j++)
                fa[mt].u[j] = *(const unsigned long long*)(&cA[base + 32 * (j >> 1) + 8 * (j & 1)]);
        }
        #pragma unroll
        for (int nt = 0; nt < 4; nt++) {
            const int base = (nbase + nt * 16 + l16) * LDSS + 16 * half;
            #pragma unroll
            for (int j = 0; j < 2; j++)
                fb[nt].q[j] = *(const int4*)(&cB[base + 32 * j]);
        }
        #pragma unroll
        for (int mt = 0; mt < 4; mt++)
            #pragma unroll
            for (int nt = 0; nt < 4; nt++)
                acc[mt][nt].v = __builtin_amdgcn_wmma_i32_16x16x64_iu8(
                    true, fa[mt].v, true, fb[nt].v, acc[mt][nt].v, false, false);
    };

    constexpr int NK = K_DIM / 64;

    issue_tile(0, 0);
    for (int kt = 0; kt < NK - 1; kt++) {
        const int cur = kt & 1;
        issue_tile(cur ^ 1, (kt + 1) * 64);
#if USE_ASYNC_COPY
        __builtin_amdgcn_s_wait_asynccnt(6);
#endif
        __syncthreads();
        compute_tile(cur);
        __syncthreads();
    }
#if USE_ASYNC_COPY
    __builtin_amdgcn_s_wait_asynccnt(0);
#endif
    __syncthreads();
    compute_tile((NK - 1) & 1);

    #pragma unroll
    for (int nt = 0; nt < 4; nt++) {
        const int col    = blockN0 + nbase + nt * 16 + l16;
        const float sc   = a_scale * wscale[col];
        const float corr = zp * (float)wrowsum[col];
        const float b_c  = bias[col];
        #pragma unroll
        for (int mt = 0; mt < 4; mt++) {
            const int row0 = blockM0 + mbase + mt * 16 + 8 * half;
            #pragma unroll
            for (int r = 0; r < 8; r++) {
                float v = sc * ((float)acc[mt][nt].s[r] - corr) + b_c;
                out[(size_t)(row0 + r) * N_DIM + col] = v;
            }
        }
    }
}

extern "C" void kernel_launch(void* const* d_in, const int* in_sizes, int n_in,
                              void* d_out, int out_size, void* d_ws, size_t ws_size,
                              hipStream_t stream) {
    const float* x    = (const float*)d_in[0];
    const float* w    = (const float*)d_in[1];
    const float* bias = (const float*)d_in[2];
    float* out = (float*)d_out;

    char* ws = (char*)d_ws;
    int8_t*   xq      = (int8_t*)ws;
    int8_t*   wq      = (int8_t*)(ws + XQ_BYTES);
    float*    wscale  = (float*)(ws + XQ_BYTES + WQ_BYTES);
    int*      wrowsum = (int*)(ws + XQ_BYTES + WQ_BYTES + 16384);
    unsigned* stats   = (unsigned*)(ws + XQ_BYTES + WQ_BYTES + 32768);

    init_stats<<<1, 1, 0, stream>>>(stats);
    minmax_kernel<<<512, 256, 0, stream>>>((const float4*)x, (M_DIM * K_DIM) / 4, stats);
    compute_params<<<1, 1, 0, stream>>>(stats);
    quant_w_kernel<<<N_DIM, 256, 0, stream>>>(w, wq, wscale, wrowsum);
    quant_x_kernel<<<1024, 256, 0, stream>>>((const float4*)x, (int*)xq, stats,
                                             (M_DIM * K_DIM) / 4);
    dim3 grid(N_DIM / 256, M_DIM / 128);
    gemm_i8_wmma<<<grid, 256, 0, stream>>>(xq, wq, wscale, wrowsum, bias, stats, out);
}
